// VectorQuantizer_1047972020422
// MI455X (gfx1250) — matrix operand order repaired and checked by reading, not compiled, not run on hardware
//
#include <hip/hip_runtime.h>
#include <hip/hip_bf16.h>
#include <math.h>

typedef __attribute__((ext_vector_type(16))) __bf16 v16bf;
typedef __attribute__((ext_vector_type(8)))  __bf16 v8bf;
typedef __attribute__((ext_vector_type(8)))  float  v8f;

#define N_ROWS   32768      // 32*32*32
#define DIM      256
#define KCODES   1024
#define ROWS_PB  128        // rows per block (16 per wave, 8 waves)
#define NBLOCKS  (N_ROWS / ROWS_PB)   // 256
#define INV_N    (1.0f / 32768.0f)
#define INV_ND   (1.0f / 8388608.0f)

// ---- workspace layout (bytes) ----
// [0,4096)        : uint32 hist[1024]
// [4096,4100)     : float accum (sum of squared error)
// [4352,8448)     : float cnorm[1024]
// [8448,+1MB)     : float ct32[1024][256]   (codebook transposed, f32)
// [+1MB,+1.5MB)   : __bf16 ctbf[1024][256]  (codebook transposed, bf16)
#define WS_ACCUM_OFF 4096
#define WS_CNORM_OFF 4352
#define WS_CT32_OFF  8448
#define WS_CTBF_OFF  (8448 + 1048576)

__global__ void vq_init(unsigned int* __restrict__ hist,
                        float* __restrict__ cnorm,
                        float* __restrict__ accum) {
    int t = threadIdx.x;
    hist[t]  = 0u;
    cnorm[t] = 0.0f;
    if (t == 0) accum[0] = 0.0f;
}

// codebook (D=256, K=1024) row-major -> ct32[k][d], ctbf[k][d], cnorm[k]
__global__ void vq_prep(const float* __restrict__ cb,
                        float* __restrict__ ct32,
                        __bf16* __restrict__ ctbf,
                        float* __restrict__ cnorm) {
    int fi = blockIdx.x * 256 + threadIdx.x;   // 0 .. 262143, coalesced read
    int d = fi >> 10;          // fi / 1024
    int k = fi & 1023;         // fi % 1024
    float v = cb[fi];
    ct32[k * DIM + d] = v;
    ctbf[k * DIM + d] = (__bf16)v;
    atomicAdd(&cnorm[k], v * v);
}

__global__ __launch_bounds__(256)
void vq_main(const float* __restrict__ xin,
             const float* __restrict__ ct32,
             const __bf16* __restrict__ ctbf,
             const float* __restrict__ cnorm,
             float* __restrict__ outp,
             unsigned int* __restrict__ hist,
             float* __restrict__ accum) {
    __shared__ __align__(16) __bf16 xs[ROWS_PB][DIM + 8];  // +8 pad: bank spread
    __shared__ float cns[KCODES];
    __shared__ int   fidx[ROWS_PB];
    __shared__ float sred[256];

    const int tid  = threadIdx.x;
    const int lane = tid & 31;
    const int wave = tid >> 5;          // 0..7, owns rows [wave*16, wave*16+16)
    const int l16  = lane & 15;
    const bool hih = (lane >= 16);
    const int rowbase = blockIdx.x * ROWS_PB;

    // ---- stage rows into LDS as bf16 (coalesced: one row per iter) + cnorm ----
    for (int it = 0; it < ROWS_PB; ++it) {
        float v = xin[(size_t)(rowbase + it) * DIM + tid];
        xs[it][tid] = (__bf16)v;
    }
    #pragma unroll
    for (int c = 0; c < 4; ++c) cns[c * 256 + tid] = cnorm[c * 256 + tid];
    __syncthreads();

    // ---- preload A tile: row m of X, full K=256, per ISA 16-bit A layout ----
    const int m  = wave * 16 + l16;
    const int kh = hih ? 8 : 0;
    v16bf a[8];
    #pragma unroll
    for (int s = 0; s < 8; ++s) {
        v8bf lo = *(const v8bf*)&xs[m][s * 32 + kh];
        v8bf hi = *(const v8bf*)&xs[m][s * 32 + 16 + kh];
        #pragma unroll
        for (int e = 0; e < 8; ++e) { a[s][e] = lo[e]; a[s][8 + e] = hi[e]; }
    }

    // ---- scan all 1024 codes in 16-wide column tiles; keep running argmin ----
    // All 8 waves stream identical B tiles -> WGP$ reuse across waves.
    float minv[8];
    int   mini[8];
    #pragma unroll
    for (int g = 0; g < 8; ++g) { minv[g] = 3.4e38f; mini[g] = 0; }

    const int koff = hih ? 16 : 0;
    for (int ctile = 0; ctile < 64; ++ctile) {
        const int n = ctile * 16 + l16;         // global code index for this lane
        const __bf16* brow = &ctbf[(size_t)n * DIM];

        // preload whole B tile first: one load clause, waits stagger over WMMAs
        v16bf b[8];
        #pragma unroll
        for (int s = 0; s < 8; ++s) {
            v8bf blo = *(const v8bf*)(brow + s * 32 + kh);
            v8bf bhi = *(const v8bf*)(brow + s * 32 + 16 + kh);
            #pragma unroll
            for (int e = 0; e < 8; ++e) { b[s][e] = blo[e]; b[s][8 + e] = bhi[e]; }
        }

        v8f c = {0.f, 0.f, 0.f, 0.f, 0.f, 0.f, 0.f, 0.f};
        #pragma unroll
        for (int s = 0; s < 8; ++s)
            c = __builtin_amdgcn_wmma_f32_16x16x32_bf16(
                    false, a[s], false, b[s], (short)0, c, false, false);

        const float cn = cns[n];
        // n strictly increases per lane -> strict '<' keeps lowest-index min
        #pragma unroll
        for (int g = 0; g < 8; ++g) {
            float val = fmaf(-2.0f, c[g], cn);  // |c|^2 - 2 x.c (argmin-equivalent)
            if (val < minv[g]) { minv[g] = val; mini[g] = n; }
        }
    }

    // ---- cross-lane argmin within each 16-lane half (C layout: N = lane&15) ----
    #pragma unroll
    for (int g = 0; g < 8; ++g) {
        float mv = minv[g]; int mi = mini[g];
        #pragma unroll
        for (int off = 8; off > 0; off >>= 1) {
            float ov = __shfl_xor(mv, off, 16);
            int   oi = __shfl_xor(mi, off, 16);
            if (ov < mv || (ov == mv && oi < mi)) { mv = ov; mi = oi; }
        }
        if (l16 == 0) {
            int r = wave * 16 + (hih ? 8 : 0) + g;   // final row index in block
            fidx[r] = mi;
            atomicAdd(&hist[mi], 1u);
        }
    }
    __syncthreads();

    // ---- emit ste (= quantized, exact f32 gather) + accumulate SSE ----
    float err = 0.0f;
    for (int it = 0; it < ROWS_PB; ++it) {
        int idx = fidx[it];
        float q  = ct32[(size_t)idx * DIM + tid];              // coalesced, L2-hot
        float xv = xin[(size_t)(rowbase + it) * DIM + tid];
        outp[(size_t)(rowbase + it) * DIM + tid] = q;
        float df = q - xv;
        err += df * df;
    }
    sred[tid] = err;
    __syncthreads();
    for (int st = 128; st > 0; st >>= 1) {
        if (tid < st) sred[tid] += sred[tid + st];
        __syncthreads();
    }
    if (tid == 0) atomicAdd(accum, sred[0]);
}

__global__ void vq_finish(const unsigned int* __restrict__ hist,
                          const float* __restrict__ accum,
                          float* __restrict__ out3) {
    __shared__ float red[1024];
    int t = threadIdx.x;
    float p = (float)hist[t] * INV_N;
    red[t] = -p * logf(p + 1e-10f);
    __syncthreads();
    for (int st = 512; st > 0; st >>= 1) {
        if (t < st) red[t] += red[t + st];
        __syncthreads();
    }
    if (t == 0) {
        float perp = expf(red[0]);
        float mse  = accum[0] * INV_ND;
        out3[0] = perp;          // perplexity
        out3[1] = mse;           // codebook_loss
        out3[2] = 0.25f * mse;   // commitment_loss
    }
}

extern "C" void kernel_launch(void* const* d_in, const int* in_sizes, int n_in,
                              void* d_out, int out_size, void* d_ws, size_t ws_size,
                              hipStream_t stream) {
    const float* xin = (const float*)d_in[0];     // (32,32,32,256) f32
    const float* cb  = (const float*)d_in[1];     // (256,1024) f32
    float* outp = (float*)d_out;                  // 8388608 ste + 3 scalars

    char* ws = (char*)d_ws;
    unsigned int* hist  = (unsigned int*)ws;
    float*  accum = (float*)(ws + WS_ACCUM_OFF);
    float*  cnorm = (float*)(ws + WS_CNORM_OFF);
    float*  ct32  = (float*)(ws + WS_CT32_OFF);
    __bf16* ctbf  = (__bf16*)(ws + WS_CTBF_OFF);

    vq_init<<<1, 1024, 0, stream>>>(hist, cnorm, accum);
    vq_prep<<<(DIM * KCODES) / 256, 256, 0, stream>>>(cb, ct32, ctbf, cnorm);
    vq_main<<<NBLOCKS, 256, 0, stream>>>(xin, ct32, ctbf, cnorm, outp, hist, accum);
    vq_finish<<<1, 1024, 0, stream>>>(hist, accum, outp + (size_t)N_ROWS * DIM);
}
